// SkipAttention_37598143709786
// MI455X (gfx1250) — matrix operand order repaired and checked by reading, not compiled, not run on hardware
//
#include <hip/hip_runtime.h>
#include <hip/hip_bf16.h>

// ---------------------------------------------------------------------------
// SkipAttention for MI455X (gfx1250, wave32, WMMA + async-to-LDS).
// Pipeline:
//   1) gemm_wt_bf16<f32 in, bf16 out>:  qkv = x @ qkv_w.T          [3072,1536]
//   2) gemm_wt_bf16:                    k_ctx = dx_ctx @ k_w.T     [4096, 512]
//   3) gemm_wt_bf16:                    v_ctx = x_ctx  @ v_w.T     [4096, 512]
//   4) skip_attn_kernel: fused flash attention; whole-frame masking; K/V
//      frames double-buffered in LDS via global_load_async_to_lds_b128
//      (ASYNCcnt) so frame i+1 DMA overlaps frame i WMMAs; DPP16 butterfly
//      softmax reductions (no LDS round-trips).
//   5) gemm_wt_bf16<bf16 in, f32+bias out>: out = attn @ proj_w.T + proj_b
// All matmuls use v_wmma_f32_16x16x32_bf16 (f32 accumulate).
// ---------------------------------------------------------------------------

typedef __bf16 bhalf_t;
typedef __attribute__((ext_vector_type(16))) __bf16 v16bf;
typedef __attribute__((ext_vector_type(8)))  float   v8f;

__device__ __forceinline__ v8f wmma_bf16(v16bf a, v16bf b, v8f c) {
  // (neg_a, A, neg_b, B, c_mod, C, reuse_a, reuse_b)
  return __builtin_amdgcn_wmma_f32_16x16x32_bf16(false, a, false, b,
                                                 (short)0, c, false, false);
}

// Async DMA of 16 bytes global -> LDS, tracked by ASYNCcnt.
__device__ __forceinline__ void async_load_b128_to_lds(unsigned lds_off,
                                                       const void* gptr) {
  asm volatile("global_load_async_to_lds_b128 %0, %1, off"
               :
               : "v"(lds_off), "v"((unsigned long long)(uintptr_t)gptr)
               : "memory");
}
__device__ __forceinline__ void wait_asynccnt0() {
  asm volatile("s_wait_asynccnt 0x0" ::: "memory");
}
__device__ __forceinline__ void wait_asynccnt8() {   // <=8 outstanding
  asm volatile("s_wait_asynccnt 0x8" ::: "memory");
}

// DPP16 butterfly helpers: reduce independently within each 16-lane row.
// xor masks {1,2,7,15} span GF(2)^4, so 4 steps give a full 16-lane reduction.
template <int CTRL>
__device__ __forceinline__ float dpp_bfly(float x) {
  return __builtin_bit_cast(
      float, __builtin_amdgcn_update_dpp(0, __builtin_bit_cast(int, x), CTRL,
                                         0xf, 0xf, true));
}
__device__ __forceinline__ float rowmax16(float v) {
  v = fmaxf(v, dpp_bfly<0xB1>(v));   // quad_perm(1,0,3,2)  : lane ^ 1
  v = fmaxf(v, dpp_bfly<0x4E>(v));   // quad_perm(2,3,0,1)  : lane ^ 2
  v = fmaxf(v, dpp_bfly<0x141>(v));  // row_half_mirror     : lane ^ 7
  v = fmaxf(v, dpp_bfly<0x140>(v));  // row_mirror          : lane ^ 15
  return v;
}
__device__ __forceinline__ float rowsum16(float v) {
  v += dpp_bfly<0xB1>(v);
  v += dpp_bfly<0x4E>(v);
  v += dpp_bfly<0x141>(v);
  v += dpp_bfly<0x140>(v);
  return v;
}

// A-fragment (16x32, MxK, bf16): lane<16 holds row=lane, k=0..15;
// lane>=16 holds row=lane-16, k=16..31.  One contiguous 16-chunk.
__device__ __forceinline__ v16bf load_a_frag_f32(const float* rowp, int half) {
  v16bf a;
  const float* p1 = rowp + half * 16;
  const float* p2 = rowp + half * 16 + 8;
#pragma unroll
  for (int e = 0; e < 8; ++e) { a[e] = (__bf16)p1[e]; a[8 + e] = (__bf16)p2[e]; }
  return a;
}
__device__ __forceinline__ v16bf load_a_frag_bf16(const bhalf_t* rowp, int half) {
  v16bf a;
  const bhalf_t* p1 = rowp + half * 16;
  const bhalf_t* p2 = rowp + half * 16 + 8;
#pragma unroll
  for (int e = 0; e < 8; ++e) { a[e] = p1[e]; a[8 + e] = p2[e]; }
  return a;
}

// B-fragment (32x16, KxN, bf16): lane<16 holds col=lane, k=0..15;
// lane>=16 holds col=lane-16, k=16..31.  One contiguous 16-chunk of the
// source row when B = (row-major [N,K] matrix)^T.
__device__ __forceinline__ v16bf load_b_frag_f32(const float* p) {
  v16bf b;
#pragma unroll
  for (int e = 0; e < 16; ++e) b[e] = (__bf16)p[e];
  return b;
}

// ---------------------------------------------------------------------------
// Generic  Out[M,N] = A[M,K] @ W[N,K]^T  (+ bias), bf16 WMMA, f32 accumulate.
// One wave -> 16x64 output tile.  128-thread block (4 waves) -> 64x64 tile.
// ---------------------------------------------------------------------------
template <bool A_BF16, bool OUT_F32_BIAS>
__global__ __launch_bounds__(128)
void gemm_wt_bf16(const void* __restrict__ Ap, const float* __restrict__ W,
                  void* __restrict__ Outp, const float* __restrict__ bias,
                  int M, int N, int K) {
  const int lane = threadIdx.x & 31;
  const int wave = threadIdx.x >> 5;
  const int half = lane >> 4;
  const int li   = lane & 15;
  const int m0 = blockIdx.y * 64 + wave * 16;
  const int n0 = blockIdx.x * 64;

  v8f acc[4];
#pragma unroll
  for (int n = 0; n < 4; ++n)
#pragma unroll
    for (int r = 0; r < 8; ++r) acc[n][r] = 0.0f;

  for (int k0 = 0; k0 < K; k0 += 32) {
    v16bf a;
    if (A_BF16)
      a = load_a_frag_bf16((const bhalf_t*)Ap + (size_t)(m0 + li) * K + k0, half);
    else
      a = load_a_frag_f32((const float*)Ap + (size_t)(m0 + li) * K + k0, half);
#pragma unroll
    for (int n = 0; n < 4; ++n) {
      v16bf bfrag =
          load_b_frag_f32(W + (size_t)(n0 + n * 16 + li) * K + k0 + half * 16);
      acc[n] = wmma_bf16(a, bfrag, acc[n]);
    }
  }

  // C/D layout: VGPR r -> row r (lanes 0-15) or row r+8 (lanes 16-31).
#pragma unroll
  for (int n = 0; n < 4; ++n)
#pragma unroll
    for (int r = 0; r < 8; ++r) {
      const size_t row = (size_t)(m0 + r + 8 * half);
      const int col = n0 + n * 16 + li;
      if (OUT_F32_BIAS)
        ((float*)Outp)[row * (size_t)N + col] = acc[n][r] + bias[col];
      else
        ((bhalf_t*)Outp)[row * (size_t)N + col] = (__bf16)acc[n][r];
    }
}

// ---------------------------------------------------------------------------
// Fused flash attention.  One 4-wave block per (b, h, query frame t0); wave w
// owns query rows t0*64 + w*16 .. +15.  All masks kill whole 64-key frames and
// depend only on (b, t0), so the live-frame set (a bitmask; bit 16 = self
// frame) is uniform across the block.  K/V frames are staged into
// double-buffered LDS via async DMA: frame i+1 streams in while frame i's
// WMMAs run (ASYNCcnt decrements in issue order, so waiting <=8 guarantees
// the older frame's 8 chunks per wave have landed).
// ---------------------------------------------------------------------------
__global__ __launch_bounds__(128)
void skip_attn_kernel(const bhalf_t* __restrict__ qkvp,   // [B*T0L, 3C] bf16
                      const bhalf_t* __restrict__ kctx,   // [B*TL, C]  bf16
                      const bhalf_t* __restrict__ vctx,   // [B*TL, C]  bf16
                      const unsigned char* __restrict__ ctx_mask,  // [B,T] bool8
                      bhalf_t* __restrict__ attn_out) {   // [B*T0L, C] bf16
  constexpr int T = 16, L = 64, C = 512, HD = 64, T0 = 12, NUM_SEEDS = 4;
  constexpr int T0L = T0 * L;     // 768
  constexpr int TL  = T * L;      // 1024
  constexpr int QKV_LD = 3 * C;   // 1536
  constexpr int PITCH = 72;       // padded LDS row pitch (bf16) to spread banks
  constexpr float QSCALE = 0.18033688011112042f;  // hd^-0.5 * log2(e)

  __shared__ __align__(16) __bf16 kbuf[2][64 * PITCH];  // K frame, dbl-buffered
  __shared__ __align__(16) __bf16 vbuf[2][64 * PITCH];  // V frame, dbl-buffered
  __shared__ __align__(16) __bf16 pbuf[4][16 * 64];     // per-wave P tile

  const int tid  = threadIdx.x;
  const int lane = tid & 31;
  const int wave = tid >> 5;
  const int half = lane >> 4;
  const int li   = lane & 15;
  const int t0 = blockIdx.x;      // query frame 0..11
  const int h  = blockIdx.y;
  const int b  = blockIdx.z;
  const int qrow0 = t0 * L + wave * 16;

  // Q A-fragments for d=0..31 / 32..63 with softmax scale folded in.
  const bhalf_t* qrow =
      qkvp + (size_t)(b * T0L + qrow0 + li) * QKV_LD + h * HD;
  v16bf aq[2];
#pragma unroll
  for (int ks = 0; ks < 2; ++ks) {
    aq[ks] = load_a_frag_bf16(qrow + ks * 32, half);
#pragma unroll
    for (int e = 0; e < 16; ++e) aq[ks][e] = (__bf16)(QSCALE * (float)aq[ks][e]);
  }

  // Live-frame bitmask (uniform across block): context frames pass ctx_mask
  // and the temporal mask; bit T = per-frame self attention, always live.
  unsigned live = 1u << T;
  for (int f = 0; f < T; ++f)
    if (ctx_mask[b * T + f] && f != t0 + NUM_SEEDS) live |= 1u << f;

  // Resolve a frame id to its K/V base pointers + row stride.
  auto frame_ptrs = [&](int f, const bhalf_t*& Kf, const bhalf_t*& Vf,
                        size_t& ld) {
    if (f == T) {
      const size_t base = (size_t)(b * T0L + t0 * L) * QKV_LD + h * HD;
      Kf = qkvp + base + C;        // k slice of qkv
      Vf = qkvp + base + 2 * C;    // v slice of qkv
      ld = QKV_LD;
    } else {
      const size_t base = (size_t)(b * TL + f * L) * C + h * HD;
      Kf = kctx + base;
      Vf = vctx + base;
      ld = C;
    }
  };
  // Stage one 64x64 bf16 K frame + V frame into LDS buffer p.
  // 512 x 16B chunks per matrix; each of 128 threads issues 4 K + 4 V chunks
  // (8 async instructions per wave -> ASYNCcnt += 8 per stage).
  auto stage = [&](int p, int f) {
    const bhalf_t *Kf, *Vf; size_t ld;
    frame_ptrs(f, Kf, Vf, ld);
#pragma unroll
    for (int i = 0; i < 4; ++i) {
      const int chunk = tid + i * 128;           // 0..511
      const int row   = chunk >> 3;              // 0..63
      const int c8    = (chunk & 7) * 8;         // bf16 col: 0,8,...,56
      async_load_b128_to_lds((unsigned)(uintptr_t)&kbuf[p][row * PITCH + c8],
                             Kf + (size_t)row * ld + c8);
      async_load_b128_to_lds((unsigned)(uintptr_t)&vbuf[p][row * PITCH + c8],
                             Vf + (size_t)row * ld + c8);
    }
  };

  float m_r[8], s_r[8];
  v8f acco[4];
#pragma unroll
  for (int r = 0; r < 8; ++r) { m_r[r] = -__builtin_inff(); s_r[r] = 0.0f; }
#pragma unroll
  for (int n = 0; n < 4; ++n)
#pragma unroll
    for (int r = 0; r < 8; ++r) acco[n][r] = 0.0f;

  unsigned rem = live;
  int fcur = __builtin_ctz(rem);
  rem &= rem - 1;
  int p = 0;
  stage(0, fcur);

  while (true) {
    const bool have_next = (rem != 0);
    int fnext = -1;
    if (have_next) {
      fnext = __builtin_ctz(rem);
      rem &= rem - 1;
      stage(p ^ 1, fnext);   // overlap next frame's DMA with this frame's math
      wait_asynccnt8();      // frame fcur's 8 chunks (older) are done
    } else {
      wait_asynccnt0();
    }
    __syncthreads();

    // ---- S = Q @ Kf^T (16 x 64); B-frags are contiguous 32B LDS reads.
    v8f s[4];
#pragma unroll
    for (int n = 0; n < 4; ++n) {
#pragma unroll
      for (int r = 0; r < 8; ++r) s[n][r] = 0.0f;
#pragma unroll
      for (int ks = 0; ks < 2; ++ks) {
        v16bf bk =
            *(const v16bf*)&kbuf[p][(n * 16 + li) * PITCH + ks * 32 + half * 16];
        s[n] = wmma_bf16(aq[ks], bk, s[n]);
      }
    }

    // ---- Online softmax (base-2 domain; scale pre-folded into Q).
    float ls[8];
#pragma unroll
    for (int r = 0; r < 8; ++r) {
      float lm = fmaxf(fmaxf(s[0][r], s[1][r]), fmaxf(s[2][r], s[3][r]));
      lm = rowmax16(lm);
      const float mn = fmaxf(m_r[r], lm);
      const float corr = __builtin_amdgcn_exp2f(m_r[r] - mn);
      m_r[r] = mn;
      s_r[r] *= corr;
#pragma unroll
      for (int n = 0; n < 4; ++n) acco[n][r] *= corr;
      float acc = 0.0f;
#pragma unroll
      for (int n = 0; n < 4; ++n) {
        const float pr = __builtin_amdgcn_exp2f(s[n][r] - mn);
        s[n][r] = pr;
        acc += pr;
      }
      ls[r] = rowsum16(acc);
      s_r[r] += ls[r];
    }

    // ---- C-layout P -> row-major LDS tile -> A-layout fragments.
#pragma unroll
    for (int n = 0; n < 4; ++n)
#pragma unroll
      for (int r = 0; r < 8; ++r)
        pbuf[wave][(r + 8 * half) * 64 + n * 16 + li] = (__bf16)s[n][r];
    __syncthreads();

    // ---- O += P @ Vf (16 x 64); V B-frags gathered from LDS (padded pitch).
#pragma unroll
    for (int ks = 0; ks < 2; ++ks) {
      v16bf ap = load_a_frag_bf16(&pbuf[wave][li * 64 + ks * 32], half);
#pragma unroll
      for (int n = 0; n < 4; ++n) {
        v16bf bv;  // B[k=key][n=d]: column-of-V gather
#pragma unroll
        for (int e = 0; e < 16; ++e)
          bv[e] = vbuf[p][(ks * 32 + half * 16 + e) * PITCH + n * 16 + li];
        acco[n] = wmma_bf16(ap, bv, acco[n]);
      }
    }
    __syncthreads();  // all waves done with buffer p before it is restaged

    if (!have_next) break;
    fcur = fnext;
    p ^= 1;
  }

  // Normalize and store [B*T0L, C] bf16 with head h at columns h*64..h*64+63.
#pragma unroll
  for (int n = 0; n < 4; ++n)
#pragma unroll
    for (int r = 0; r < 8; ++r) {
      const float o = acco[n][r] / s_r[r];
      const size_t row = (size_t)(b * T0L + qrow0 + r + 8 * half);
      const int col = h * HD + n * 16 + li;
      attn_out[row * C + col] = (__bf16)o;
    }
}

// ---------------------------------------------------------------------------
extern "C" void kernel_launch(void* const* d_in, const int* in_sizes, int n_in,
                              void* d_out, int out_size, void* d_ws, size_t ws_size,
                              hipStream_t stream) {
  (void)in_sizes; (void)n_in; (void)out_size; (void)ws_size;

  const float* x      = (const float*)d_in[0];
  const float* x_ctx  = (const float*)d_in[1];
  const float* dx_ctx = (const float*)d_in[2];
  const unsigned char* ctx_mask = (const unsigned char*)d_in[3];  // bool8 [B,T]
  const float* qkv_w  = (const float*)d_in[4];
  const float* k_w    = (const float*)d_in[5];
  const float* v_w    = (const float*)d_in[6];
  const float* proj_w = (const float*)d_in[7];
  const float* proj_b = (const float*)d_in[8];

  constexpr int B = 4, T = 16, L = 64, C = 512, T0 = 12;
  constexpr int M_Q = B * T0 * L;  // 3072 query rows
  constexpr int M_C = B * T * L;   // 4096 context rows

  // bf16 workspace (~20 MB total).
  bhalf_t* qkvp = (bhalf_t*)d_ws;                      // [3072, 1536]
  bhalf_t* kctx = qkvp + (size_t)M_Q * 3 * C;          // [4096,  512]
  bhalf_t* vctx = kctx + (size_t)M_C * C;              // [4096,  512]
  bhalf_t* attn = vctx + (size_t)M_C * C;              // [3072,  512]

  const dim3 blk(128);
  gemm_wt_bf16<false, false><<<dim3((3 * C) / 64, M_Q / 64), blk, 0, stream>>>(
      x, qkv_w, qkvp, nullptr, M_Q, 3 * C, C);
  gemm_wt_bf16<false, false><<<dim3(C / 64, M_C / 64), blk, 0, stream>>>(
      dx_ctx, k_w, kctx, nullptr, M_C, C, C);
  gemm_wt_bf16<false, false><<<dim3(C / 64, M_C / 64), blk, 0, stream>>>(
      x_ctx, v_w, vctx, nullptr, M_C, C, C);

  skip_attn_kernel<<<dim3(T0, 8, B), dim3(128), 0, stream>>>(
      qkvp, kctx, vctx, ctx_mask, attn);

  gemm_wt_bf16<true, true><<<dim3(C / 64, M_Q / 64), blk, 0, stream>>>(
      attn, proj_w, (float*)d_out, proj_b, M_Q, C, C);
}
